// GAT_17523466568164
// MI455X (gfx1250) — compile-verified, matrix operand order checked by reading, not run on hardware
//
#include <hip/hip_runtime.h>
#include <hip/hip_bf16.h>
#include <math.h>

// ---------------- problem constants (match reference) ----------------
#define N_NODES 10000
#define N_EDGES 80000
#define E_TOT   (N_EDGES + N_NODES)   // reference appends self loops
#define F_IN    128
#define HID     300
#define H1      5
#define H3      3
#define N_CLS   40
#define HC1     (H1 * HID)            // 1500
#define HC3     (H3 * N_CLS)          // 120
#define NEG_SLOPE 0.2f
#define EPS_F     1e-16f

// padded GEMM dims (M,N -> mult of 128, K -> mult of 32)
#define MPAD      10112               // 79 * 128
#define KPAD1     128                 // F_IN already mult of 32
#define KPAD23    1536                // 1500 -> 1536
#define NPAD12    1536                // 1500 -> 1536 (12 * 128)
#define NPAD3     128                 // 120  -> 128  (1 * 128)

// ---------------- WMMA vector types ----------------
typedef __bf16 v16bf __attribute__((ext_vector_type(16)));
typedef float  v8f   __attribute__((ext_vector_type(8)));

// f32 -> bf16 bits (RNE)
__device__ __forceinline__ unsigned int bfbits(float f) {
    unsigned int u = __float_as_uint(f);
    return ((u + 0x7FFFu + ((u >> 16) & 1u)) >> 16) & 0xFFFFu;
}
__device__ __forceinline__ unsigned int pack2bf(float a, float b) {
    return bfbits(a) | (bfbits(b) << 16);
}

// monotonic float<->uint mapping for atomicMax-based segment max
__device__ __forceinline__ unsigned int f2ord(float f) {
    unsigned int u = __float_as_uint(f);
    return (u & 0x80000000u) ? ~u : (u | 0x80000000u);
}
__device__ __forceinline__ float ord2f(unsigned int o) {
    unsigned int u = (o & 0x80000000u) ? (o & 0x7FFFFFFFu) : ~o;
    return __uint_as_float(u);
}
#define ORD_NEG_INF 0x007FFFFFu   // f2ord(-inf)

// =====================================================================
// Conversion / padding kernels (bandwidth-trivial vs the GEMM)
// =====================================================================
__global__ void conv_A_bf16(const float* __restrict__ A, int M, int K,
                            unsigned int* __restrict__ out, int Kpad, int total)
{
    int i = blockIdx.x * blockDim.x + threadIdx.x;
    if (i >= total) return;
    int kh = Kpad >> 1;
    int r = i / kh, k = (i - r * kh) * 2;
    float v0 = 0.f, v1 = 0.f;
    if (r < M) {
        if (k + 1 < K) {
            float2 t = *(const float2*)(A + (size_t)r * K + k);
            v0 = t.x; v1 = t.y;
        } else if (k < K) {
            v0 = A[(size_t)r * K + k];
        }
    }
    out[i] = pack2bf(v0, v1);
}

// B f32 [K x N] -> bf16 transposed [Npad x Kpad], zero-padded
__global__ void conv_Bt_bf16(const float* __restrict__ B, int K, int N,
                             unsigned int* __restrict__ out, int Kpad, int total)
{
    int i = blockIdx.x * blockDim.x + threadIdx.x;
    if (i >= total) return;
    int kh = Kpad >> 1;
    int n = i / kh, k = (i - n * kh) * 2;
    float v0 = 0.f, v1 = 0.f;
    if (n < N) {
        if (k < K)     v0 = B[(size_t)k * N + n];
        if (k + 1 < K) v1 = B[(size_t)(k + 1) * N + n];
    }
    out[i] = pack2bf(v0, v1);
}

// =====================================================================
// GEMM: C[M,N] = A * B from pre-padded bf16 buffers.
//   Abf : [MPAD x Kpad] row-major bf16
//   Btbf: [Npad x Kpad] bf16 (B transposed -> K contiguous)
// Block 256 = 8 waves (4 x 2); macro-tile 128(M) x 128(N);
// each wave owns a 32x64 tile = 2x4 accumulators = 8 WMMAs / K-step.
// A fragments reused 4x, B fragments 2x. Inner loop is branchless.
// =====================================================================
__global__ __launch_bounds__(256) void gemm_wmma_bf16(
    const unsigned short* __restrict__ Abf,
    const unsigned short* __restrict__ Btbf,
    float* __restrict__ C, int ldc, int M, int N, int Kpad)
{
    // uint = 2 bf16 along K; 16 uints = 32 K per row; +1 pad vs bank conflicts
    __shared__ unsigned int As[128][17];
    __shared__ unsigned int Bs[128][17];

    const int tid  = threadIdx.x;
    const int lane = tid & 31;
    const int w    = tid >> 5;
    const int wm   = w >> 1;                 // 0..3  -> M offset wm*32
    const int wn   = w & 1;                  // 0..1  -> N offset wn*64
    const int m0   = blockIdx.y * 128;
    const int n0   = blockIdx.x * 128;

    const int lm = lane & 15;                // M/N index within 16-tile
    const int ku = (lane >> 4) * 4;          // uint offset for lane group

    v8f acc[2][4];
    #pragma unroll
    for (int i = 0; i < 2; ++i)
        #pragma unroll
        for (int j = 0; j < 4; ++j)
            acc[i][j] = (v8f){0.f,0.f,0.f,0.f,0.f,0.f,0.f,0.f};

    // loader mapping (fixed, branchless): 2 uint4 each for A and B
    const int rr = tid >> 2;                 // rows 0..63 (first half)
    const int qq = tid & 3;                  // which uint4 of the 32-K row

    for (int k0 = 0; k0 < Kpad; k0 += 32) {
        // -------- branchless global loads (b128) --------
        const uint4 va0 = *((const uint4*)(Abf  + (size_t)(m0 + rr)      * Kpad + k0) + qq);
        const uint4 va1 = *((const uint4*)(Abf  + (size_t)(m0 + rr + 64) * Kpad + k0) + qq);
        const uint4 vb0 = *((const uint4*)(Btbf + (size_t)(n0 + rr)      * Kpad + k0) + qq);
        const uint4 vb1 = *((const uint4*)(Btbf + (size_t)(n0 + rr + 64) * Kpad + k0) + qq);

        // speculative prefetch of next K tile (stays inside padded buffers)
        if (k0 + 32 < Kpad) {
            __builtin_prefetch(Abf  + (size_t)(m0 + (tid >> 1)) * Kpad + (k0 + 32), 0, 3);
            __builtin_prefetch(Btbf + (size_t)(n0 + (tid >> 1)) * Kpad + (k0 + 32), 0, 3);
        }

        __syncthreads();   // previous tile fully consumed
        {
            const int c = qq * 4;
            As[rr]     [c]     = va0.x; As[rr]     [c + 1] = va0.y;
            As[rr]     [c + 2] = va0.z; As[rr]     [c + 3] = va0.w;
            As[rr + 64][c]     = va1.x; As[rr + 64][c + 1] = va1.y;
            As[rr + 64][c + 2] = va1.z; As[rr + 64][c + 3] = va1.w;
            Bs[rr]     [c]     = vb0.x; Bs[rr]     [c + 1] = vb0.y;
            Bs[rr]     [c + 2] = vb0.z; Bs[rr]     [c + 3] = vb0.w;
            Bs[rr + 64][c]     = vb1.x; Bs[rr + 64][c + 1] = vb1.y;
            Bs[rr + 64][c + 2] = vb1.z; Bs[rr + 64][c + 3] = vb1.w;
        }
        __syncthreads();

        // -------- fragments per ISA 16-bit A/B layouts --------
        union Frag { unsigned int u[8]; v16bf v; };
        Frag a[2], b[4];
        #pragma unroll
        for (int t = 0; t < 2; ++t) {
            const int ar = wm * 32 + t * 16 + lm;
            #pragma unroll
            for (int j = 0; j < 4; ++j) {
                a[t].u[j]     = As[ar][ku + j];
                a[t].u[4 + j] = As[ar][ku + 8 + j];
            }
        }
        #pragma unroll
        for (int t = 0; t < 4; ++t) {
            const int bn = wn * 64 + t * 16 + lm;
            #pragma unroll
            for (int j = 0; j < 4; ++j) {
                b[t].u[j]     = Bs[bn][ku + j];
                b[t].u[4 + j] = Bs[bn][ku + 8 + j];
            }
        }
        #pragma unroll
        for (int i = 0; i < 2; ++i)
            #pragma unroll
            for (int j = 0; j < 4; ++j)
                acc[i][j] = __builtin_amdgcn_wmma_f32_16x16x32_bf16(
                    false, a[i].v, false, b[j].v, (short)0, acc[i][j], false, false);
    }

    // -------- epilogue: store 8 tiles per 16x16 f32 C/D layout --------
    const int rbase = m0 + wm * 32 + 8 * (lane >> 4);
    #pragma unroll
    for (int i = 0; i < 2; ++i) {
        #pragma unroll
        for (int j = 0; j < 4; ++j) {
            const int col = n0 + wn * 64 + j * 16 + lm;
            if (col >= N) continue;
            #pragma unroll
            for (int r = 0; r < 8; ++r) {
                int row = rbase + i * 16 + r;
                if (row < M) C[(size_t)row * ldc + col] = acc[i][j][r];
            }
        }
    }
}

// =====================================================================
// Attention pipeline kernels
// =====================================================================
__global__ void fill_f32(float* p, float v, int n) {
    int i = blockIdx.x * blockDim.x + threadIdx.x;
    if (i < n) p[i] = v;
}
__global__ void fill_u32(unsigned int* p, unsigned int v, int n) {
    int i = blockIdx.x * blockDim.x + threadIdx.x;
    if (i < n) p[i] = v;
}

// per (node, head) attention dot products: a_src[n,h] / a_dst[n,h]
__global__ void att_dots(const float* __restrict__ hlin,
                         const float* __restrict__ asv,
                         const float* __restrict__ adv,
                         float* __restrict__ asrc, float* __restrict__ adst,
                         int Nn, int H, int C)
{
    int i = blockIdx.x * blockDim.x + threadIdx.x;
    if (i >= Nn * H) return;
    int n = i / H, h = i - n * H;
    const float* hp = hlin + (size_t)n * H * C + (size_t)h * C;
    const float* sp = asv + h * C;
    const float* dp = adv + h * C;
    float s = 0.f, d = 0.f;
    for (int c = 0; c < C; ++c) {
        float v = hp[c];
        s += v * sp[c];
        d += v * dp[c];
    }
    asrc[i] = s;
    adst[i] = d;
}

// e = leaky_relu(a_src[src] + a_dst[dst]); segment max via ordered atomicMax
__global__ void edge_pass1(const int* __restrict__ ei,
                           const float* __restrict__ asrc,
                           const float* __restrict__ adst,
                           float* __restrict__ ebuf,
                           unsigned int* __restrict__ mord, int H)
{
    int idx = blockIdx.x * blockDim.x + threadIdx.x;
    if (idx >= E_TOT * H) return;
    int e = idx / H, h = idx - e * H;
    int s, d;
    if (e < N_EDGES) { s = ei[e]; d = ei[N_EDGES + e]; }
    else             { s = d = e - N_EDGES; }
    float v = asrc[s * H + h] + adst[d * H + h];
    v = (v > 0.f) ? v : NEG_SLOPE * v;
    ebuf[idx] = v;
    atomicMax(&mord[d * H + h], f2ord(v));
}

// ex = exp(e - m[dst]); segment sum via atomicAdd
__global__ void edge_pass2(const int* __restrict__ ei,
                           const unsigned int* __restrict__ mord,
                           float* __restrict__ ebuf,
                           float* __restrict__ ssum, int H)
{
    int idx = blockIdx.x * blockDim.x + threadIdx.x;
    if (idx >= E_TOT * H) return;
    int e = idx / H, h = idx - e * H;
    int d = (e < N_EDGES) ? ei[N_EDGES + e] : (e - N_EDGES);
    float m  = ord2f(mord[d * H + h]);
    float ex = expf(ebuf[idx] - m);
    ebuf[idx] = ex;
    atomicAdd(&ssum[d * H + h], ex);
}

// out[dst] += alpha * h[src] ; one block per edge, threads over H*C channels
__global__ void edge_aggregate(const int* __restrict__ ei,
                               const float* __restrict__ ex,
                               const float* __restrict__ ssum,
                               const float* __restrict__ hlin,
                               float* __restrict__ outp, int H, int C)
{
    int e = blockIdx.x;
    int s, d;
    if (e < N_EDGES) { s = ei[e]; d = ei[N_EDGES + e]; }
    else             { s = d = e - N_EDGES; }
    int HC = H * C;
    for (int c = threadIdx.x; c < HC; c += blockDim.x) {
        int h = c / C;
        float alpha = ex[e * H + h] / (ssum[d * H + h] + EPS_F);
        atomicAdd(&outp[(size_t)d * HC + c], alpha * hlin[(size_t)s * HC + c]);
    }
}

// act = elu(agg + bias (+ resid))   (resid may alias act: same-index rw)
__global__ void finalize_elu(const float* __restrict__ agg,
                             const float* __restrict__ bias,
                             const float* __restrict__ resid,
                             float* __restrict__ outp, int total, int HC)
{
    int i = blockIdx.x * blockDim.x + threadIdx.x;
    if (i >= total) return;
    int c = i % HC;
    float v = agg[i] + bias[c];
    if (resid) v += resid[i];
    outp[i] = (v > 0.f) ? v : expm1f(v);
}

// layer 3: mean over heads + bias, then global sum-pool via atomicAdd
__global__ void layer3_pool(const float* __restrict__ agg,
                            const float* __restrict__ b3,
                            float* __restrict__ pooled, int Nn, int H, int C)
{
    int i = blockIdx.x * blockDim.x + threadIdx.x;
    if (i >= Nn * C) return;
    int n = i / C, c = i - n * C;
    float s = 0.f;
    for (int h = 0; h < H; ++h) s += agg[(size_t)n * H * C + (size_t)h * C + c];
    s = s / (float)H + b3[c];
    atomicAdd(&pooled[c], s);
}

// out[0..C-1] = pooled ; out[C..2C-1] = log_softmax(pooled)
__global__ void final_logsoftmax(const float* __restrict__ pooled,
                                 float* __restrict__ outp, int C)
{
    if (blockIdx.x == 0 && threadIdx.x == 0) {
        float mx = -INFINITY;
        for (int c = 0; c < C; ++c) mx = fmaxf(mx, pooled[c]);
        float s = 0.f;
        for (int c = 0; c < C; ++c) s += expf(pooled[c] - mx);
        float l = logf(s);
        for (int c = 0; c < C; ++c) {
            outp[c]     = pooled[c];
            outp[C + c] = pooled[c] - mx - l;
        }
    }
}

// =====================================================================
// Host driver
// =====================================================================
extern "C" void kernel_launch(void* const* d_in, const int* in_sizes, int n_in,
                              void* d_out, int out_size, void* d_ws, size_t ws_size,
                              hipStream_t stream) {
    const float* x   = (const float*)d_in[0];
    const int*   ei  = (const int*)  d_in[1];
    const float* W1  = (const float*)d_in[2];
    const float* as1 = (const float*)d_in[3];
    const float* ad1 = (const float*)d_in[4];
    const float* b1  = (const float*)d_in[5];
    const float* W2  = (const float*)d_in[6];
    const float* as2 = (const float*)d_in[7];
    const float* ad2 = (const float*)d_in[8];
    const float* b2  = (const float*)d_in[9];
    const float* W3  = (const float*)d_in[10];
    const float* as3 = (const float*)d_in[11];
    const float* ad3 = (const float*)d_in[12];
    const float* b3  = (const float*)d_in[13];
    float* out = (float*)d_out;

    // ---- workspace carve ----
    char* w = (char*)d_ws;
    size_t off = 0;
    auto carve = [&](size_t bytes) -> void* {
        void* p = w + off;
        off = (off + bytes + 255) & ~(size_t)255;
        return p;
    };
    float*          hlin   = (float*)         carve((size_t)N_NODES * HC1 * 4);
    float*          agg    = (float*)         carve((size_t)N_NODES * HC1 * 4);
    float*          act    = (float*)         carve((size_t)N_NODES * HC1 * 4);
    unsigned short* Abf    = (unsigned short*)carve((size_t)MPAD * KPAD23 * 2);
    unsigned short* Btbf   = (unsigned short*)carve((size_t)NPAD12 * KPAD23 * 2);
    float*          asrc   = (float*)         carve((size_t)N_NODES * H1 * 4);
    float*          adst   = (float*)         carve((size_t)N_NODES * H1 * 4);
    unsigned int*   mord   = (unsigned int*)  carve((size_t)N_NODES * H1 * 4);
    float*          ssum   = (float*)         carve((size_t)N_NODES * H1 * 4);
    float*          ebuf   = (float*)         carve((size_t)E_TOT * H1 * 4);
    float*          pooled = (float*)         carve(256);

    auto run_layer = [&](const float* in_act, int K, int Kpad,
                         const float* Wm, int Npad,
                         const float* asv, const float* adv, int H, int C) {
        int HC = H * C;
        // pad + convert A and W^T to bf16
        int totA = MPAD * (Kpad / 2);
        conv_A_bf16<<<(totA + 255) / 256, 256, 0, stream>>>(
            in_act, N_NODES, K, (unsigned int*)Abf, Kpad, totA);
        int totB = Npad * (Kpad / 2);
        conv_Bt_bf16<<<(totB + 255) / 256, 256, 0, stream>>>(
            Wm, K, HC, (unsigned int*)Btbf, Kpad, totB);
        // dense projection via WMMA
        dim3 gg(Npad / 128, MPAD / 128);
        gemm_wmma_bf16<<<gg, 256, 0, stream>>>(Abf, Btbf, hlin, HC,
                                               N_NODES, HC, Kpad);
        // attention scalars
        att_dots<<<(N_NODES * H + 255) / 256, 256, 0, stream>>>(
            hlin, asv, adv, asrc, adst, N_NODES, H, C);
        fill_u32<<<(N_NODES * H + 255) / 256, 256, 0, stream>>>(
            mord, ORD_NEG_INF, N_NODES * H);
        fill_f32<<<(N_NODES * H + 255) / 256, 256, 0, stream>>>(
            ssum, 0.f, N_NODES * H);
        // segment softmax over incoming edges
        edge_pass1<<<(E_TOT * H + 255) / 256, 256, 0, stream>>>(
            ei, asrc, adst, ebuf, mord, H);
        edge_pass2<<<(E_TOT * H + 255) / 256, 256, 0, stream>>>(
            ei, mord, ebuf, ssum, H);
        // weighted aggregation
        fill_f32<<<(N_NODES * HC + 255) / 256, 256, 0, stream>>>(
            agg, 0.f, N_NODES * HC);
        edge_aggregate<<<E_TOT, 256, 0, stream>>>(ei, ebuf, ssum, hlin, agg, H, C);
    };

    // ---- layer 1: x -> act = elu(gat1(x) + b1) ----
    run_layer(x, F_IN, KPAD1, W1, NPAD12, as1, ad1, H1, HID);
    finalize_elu<<<(N_NODES * HC1 + 255) / 256, 256, 0, stream>>>(
        agg, b1, nullptr, act, N_NODES * HC1, HC1);

    // ---- layer 2: act = elu(act + gat2(act) + b2) (residual, in place) ----
    run_layer(act, HC1, KPAD23, W2, NPAD12, as2, ad2, H1, HID);
    finalize_elu<<<(N_NODES * HC1 + 255) / 256, 256, 0, stream>>>(
        agg, b2, act, act, N_NODES * HC1, HC1);

    // ---- layer 3: head-mean + bias, global sum pool, log-softmax ----
    run_layer(act, HC1, KPAD23, W3, NPAD3, as3, ad3, H3, N_CLS);
    fill_f32<<<1, 64, 0, stream>>>(pooled, 0.f, 64);
    layer3_pool<<<(N_NODES * N_CLS + 255) / 256, 256, 0, stream>>>(
        agg, b3, pooled, N_NODES, H3, N_CLS);
    final_logsoftmax<<<1, 32, 0, stream>>>(pooled, out, N_CLS);
}
